// ModuleBatchedExperts_21157008900422
// MI455X (gfx1250) — matrix operand order repaired and checked by reading, not compiled, not run on hardware
//
#include <hip/hip_runtime.h>
#include <hip/hip_bf16.h>

// Problem constants (from reference)
constexpr int NT = 4096;   // tokens
constexpr int DD = 1024;   // model dim
constexpr int NE = 8;      // experts
constexpr int HH = 2048;   // hidden dim
constexpr int TK = 2;      // top-k

typedef __attribute__((ext_vector_type(16))) __bf16 v16bf;
typedef __attribute__((ext_vector_type(8)))  __bf16 v8bf;
typedef __attribute__((ext_vector_type(8)))  float  v8f;
typedef __attribute__((ext_vector_type(4)))  int    v4i;

// ---- workspace layout (bytes) ----
constexpr size_t wsalign(size_t x) { return (x + 255) & ~(size_t)255; }
constexpr size_t OFF_CNT  = 0;                                        // [NE] int
constexpr size_t OFF_OFFS = 256;                                      // [NE] int
constexpr size_t OFF_LIST = 512;                                      // [NE][NT] int
constexpr size_t OFF_WT   = OFF_LIST + (size_t)NE * NT * 4;           // [NE][NT] f32
constexpr size_t OFF_SLOT = OFF_WT   + (size_t)NE * NT * 4;           // [NE][NT] int
constexpr size_t OFF_XB   = wsalign(OFF_SLOT + (size_t)NE * NT * 4);  // [NT][DD] bf16
constexpr size_t OFF_W1T  = wsalign(OFF_XB  + (size_t)NT * DD * 2);   // [NE][HH][DD] bf16
constexpr size_t OFF_W2T  = wsalign(OFF_W1T + (size_t)NE * DD * HH * 2); // [NE][DD][HH] bf16
constexpr size_t OFF_HB   = wsalign(OFF_W2T + (size_t)NE * DD * HH * 2); // [NT*TK][HH] bf16
constexpr size_t OFF_YC   = wsalign(OFF_HB  + (size_t)NT * TK * HH * 2); // [NT][TK][DD] f32

__device__ __forceinline__ float gelu_exact(float x) {
    return 0.5f * x * (1.0f + erff(x * 0.70710678118654752440f));
}

// ---- CDNA5 async global->LDS copy (ASYNCcnt path), with safe fallback ----
#if __has_builtin(__builtin_amdgcn_global_load_async_to_lds_b128)
#define HAS_ASYNC_LDS 1
#else
#define HAS_ASYNC_LDS 0
#endif

__device__ __forceinline__ void async_b128(const __bf16* g, __bf16* l) {
#if HAS_ASYNC_LDS
    __builtin_amdgcn_global_load_async_to_lds_b128(
        (__attribute__((address_space(1))) v4i*)g,
        (__attribute__((address_space(3))) v4i*)l, 0, 0);
#else
    *(v8bf*)l = *(const v8bf*)g;   // sync fallback: global_load_b128 + ds_store_b128
#endif
}

template <int N>
__device__ __forceinline__ void wait_async() {
#if HAS_ASYNC_LDS
#if __has_builtin(__builtin_amdgcn_s_wait_asynccnt)
    __builtin_amdgcn_s_wait_asynccnt(N);
#else
    asm volatile("s_wait_asynccnt %0" :: "i"(N) : "memory");
#endif
#endif
}

__device__ __forceinline__ v16bf frag16(const __bf16* p) {
    v8bf lo = *(const v8bf*)(p);
    v8bf hi = *(const v8bf*)(p + 16);
    return __builtin_shufflevector(lo, hi, 0,1,2,3,4,5,6,7,8,9,10,11,12,13,14,15);
}

// ---------------- routing compaction ----------------
__global__ void k_zero_counts(int* cnt) {
    if (threadIdx.x < NE) cnt[threadIdx.x] = 0;
}

__global__ void k_route(const float* __restrict__ routing, int* cnt,
                        int* __restrict__ list, float* __restrict__ wt,
                        int* __restrict__ slot) {
    int n = blockIdx.x * blockDim.x + threadIdx.x;
    if (n >= NT) return;
    int s = 0;
    for (int e = 0; e < NE; ++e) {
        float r = routing[n * NE + e];
        if (r > 0.0f && s < TK) {
            int p = atomicAdd(&cnt[e], 1);
            list[e * NT + p] = n;
            wt[e * NT + p]   = r;
            slot[e * NT + p] = s;
            ++s;
        }
    }
}

__global__ void k_offsets(const int* __restrict__ cnt, int* __restrict__ offs) {
    if (threadIdx.x == 0) {
        int o = 0;
        for (int e = 0; e < NE; ++e) { offs[e] = o; o += cnt[e]; }
    }
}

// ---------------- precision / layout conversion ----------------
__global__ void k_cvt_x(const float* __restrict__ x, __bf16* __restrict__ xb) {
    size_t i = (size_t)blockIdx.x * blockDim.x + threadIdx.x;
    if (i < (size_t)NT * DD) xb[i] = (__bf16)x[i];
}

// W1 [E,D,H] -> W1t [E,H,D] bf16 (B of GEMM1, K-contiguous per output column)
__global__ void k_t_w1(const float* __restrict__ W1, __bf16* __restrict__ W1t) {
    size_t i = (size_t)blockIdx.x * blockDim.x + threadIdx.x;
    if (i >= (size_t)NE * DD * HH) return;
    size_t e = i / ((size_t)DD * HH);
    size_t rem = i % ((size_t)DD * HH);
    size_t h = rem / DD, d = rem % DD;
    W1t[i] = (__bf16)W1[e * (size_t)DD * HH + d * HH + h];
}

// W2 [E,H,D] -> W2t [E,D,H] bf16 (B of GEMM2)
__global__ void k_t_w2(const float* __restrict__ W2, __bf16* __restrict__ W2t) {
    size_t i = (size_t)blockIdx.x * blockDim.x + threadIdx.x;
    if (i >= (size_t)NE * DD * HH) return;
    size_t e = i / ((size_t)DD * HH);
    size_t rem = i % ((size_t)DD * HH);
    size_t d = rem / HH, h = rem % HH;
    W2t[i] = (__bf16)W2[e * (size_t)DD * HH + h * DD + d];
}

__global__ void k_zero_f32(float* __restrict__ p, size_t n) {
    size_t i = (size_t)blockIdx.x * blockDim.x + threadIdx.x;
    if (i < n) p[i] = 0.0f;
}

// ---------------- GEMM1: h = gelu(Xg @ W1[e] + b1[e]) ----------------
// block = 256 threads (8 waves); block tile M=32, N=512; wave tile 32x64.
// A (gathered token rows) staged through LDS with double-buffered async copies.
__global__ __launch_bounds__(256) void k_gemm1(
    const __bf16* __restrict__ xb,    // [NT,DD]
    const __bf16* __restrict__ w1t,   // [NE,HH,DD]
    const float*  __restrict__ b1,    // [NE,HH]
    const int* __restrict__ cnt, const int* __restrict__ offs,
    const int* __restrict__ list,
    __bf16* __restrict__ hb)          // compact [sum cnt, HH]
{
    const int e  = blockIdx.z;
    const int c  = cnt[e];
    const int m0 = blockIdx.y << 5;        // 32 rows / block
    if (m0 >= c) return;

    __shared__ alignas(16) __bf16 Abuf[2][32 * 64];

    const int tid  = threadIdx.x;
    const int lane = tid & 31;
    const int wave = tid >> 5;
    const int grp  = lane >> 4;
    const int lr   = lane & 15;
    const int n0   = blockIdx.x * 512 + wave * 64;

    // producer mapping: 256 threads x one b128 = full 32x64 bf16 tile per stage
    const int prow = tid >> 3;             // 0..31
    const int pk   = (tid & 7) * 8;        // 0,8,...,56
    int prg = m0 + prow; if (prg >= c) prg = c - 1;
    const __bf16* __restrict__ psrc = xb + (size_t)list[e * NT + prg] * DD + pk;
    __bf16* const pdst0 = &Abuf[0][prow * 64 + pk];
    __bf16* const pdst1 = &Abuf[1][prow * 64 + pk];

    const __bf16* __restrict__ Bcol = w1t + ((size_t)e * HH + (size_t)(n0 + lr)) * DD;

    v8f acc[8] = {};
    async_b128(psrc, pdst0);

    int cur = 0;
    for (int k0 = 0; k0 < DD; k0 += 64) {
        if (k0 + 64 < DD) {
            async_b128(psrc + k0 + 64, cur ? pdst0 : pdst1);
            wait_async<1>();
        } else {
            wait_async<0>();
        }
        __syncthreads();
        const __bf16* As = &Abuf[cur][0];
#pragma unroll
        for (int kk = 0; kk < 64; kk += 32) {
            v16bf a[2];
#pragma unroll
            for (int s = 0; s < 2; ++s)
                a[s] = frag16(As + (s * 16 + lr) * 64 + kk + grp * 8);
#pragma unroll
            for (int t = 0; t < 4; ++t) {
                const v16bf b = frag16(Bcol + (size_t)t * 16 * DD + k0 + kk + grp * 8);
                acc[t * 2 + 0] = __builtin_amdgcn_wmma_f32_16x16x32_bf16(
                    false, a[0], false, b, (short)0, acc[t * 2 + 0], false, false);
                acc[t * 2 + 1] = __builtin_amdgcn_wmma_f32_16x16x32_bf16(
                    false, a[1], false, b, (short)0, acc[t * 2 + 1], false, false);
            }
        }
        __syncthreads();
        cur ^= 1;
    }

    const size_t hbase = (size_t)offs[e];
#pragma unroll
    for (int t = 0; t < 4; ++t) {
        const int colg = n0 + t * 16 + lr;
        const float bias = b1[e * HH + colg];
#pragma unroll
        for (int s = 0; s < 2; ++s) {
#pragma unroll
            for (int r = 0; r < 8; ++r) {
                const int m = m0 + s * 16 + r + grp * 8;
                if (m < c) {
                    const float v = gelu_exact(acc[t * 2 + s][r] + bias);
                    hb[(hbase + m) * (size_t)HH + colg] = (__bf16)v;
                }
            }
        }
    }
}

// ---------------- GEMM2: yc[token,slot] = (h @ W2[e] + b2[e]) * wt ----------------
__global__ __launch_bounds__(256) void k_gemm2(
    const __bf16* __restrict__ hb,    // compact [sum cnt, HH]
    const __bf16* __restrict__ w2t,   // [NE,DD,HH]
    const float*  __restrict__ b2,    // [NE,DD]
    const int* __restrict__ cnt, const int* __restrict__ offs,
    const int* __restrict__ list, const float* __restrict__ wt,
    const int* __restrict__ slot,
    float* __restrict__ yc)           // [NT,TK,DD]
{
    const int e  = blockIdx.z;
    const int c  = cnt[e];
    const int m0 = blockIdx.y << 5;
    if (m0 >= c) return;

    __shared__ alignas(16) __bf16 Abuf[2][32 * 64];

    const int tid  = threadIdx.x;
    const int lane = tid & 31;
    const int wave = tid >> 5;
    const int grp  = lane >> 4;
    const int lr   = lane & 15;
    const int n0   = blockIdx.x * 512 + wave * 64;
    const size_t hbase = (size_t)offs[e];

    const int prow = tid >> 3;
    const int pk   = (tid & 7) * 8;
    int prg = m0 + prow; if (prg >= c) prg = c - 1;
    const __bf16* __restrict__ psrc = hb + (hbase + (size_t)prg) * HH + pk;
    __bf16* const pdst0 = &Abuf[0][prow * 64 + pk];
    __bf16* const pdst1 = &Abuf[1][prow * 64 + pk];

    const __bf16* __restrict__ Bcol = w2t + ((size_t)e * DD + (size_t)(n0 + lr)) * HH;

    v8f acc[8] = {};
    async_b128(psrc, pdst0);

    int cur = 0;
    for (int k0 = 0; k0 < HH; k0 += 64) {
        if (k0 + 64 < HH) {
            async_b128(psrc + k0 + 64, cur ? pdst0 : pdst1);
            wait_async<1>();
        } else {
            wait_async<0>();
        }
        __syncthreads();
        const __bf16* As = &Abuf[cur][0];
#pragma unroll
        for (int kk = 0; kk < 64; kk += 32) {
            v16bf a[2];
#pragma unroll
            for (int s = 0; s < 2; ++s)
                a[s] = frag16(As + (s * 16 + lr) * 64 + kk + grp * 8);
#pragma unroll
            for (int t = 0; t < 4; ++t) {
                const v16bf b = frag16(Bcol + (size_t)t * 16 * HH + k0 + kk + grp * 8);
                acc[t * 2 + 0] = __builtin_amdgcn_wmma_f32_16x16x32_bf16(
                    false, a[0], false, b, (short)0, acc[t * 2 + 0], false, false);
                acc[t * 2 + 1] = __builtin_amdgcn_wmma_f32_16x16x32_bf16(
                    false, a[1], false, b, (short)0, acc[t * 2 + 1], false, false);
            }
        }
        __syncthreads();
        cur ^= 1;
    }

#pragma unroll
    for (int t = 0; t < 4; ++t) {
        const int colg = n0 + t * 16 + lr;
        const float bias = b2[e * DD + colg];
#pragma unroll
        for (int s = 0; s < 2; ++s) {
#pragma unroll
            for (int r = 0; r < 8; ++r) {
                const int m = m0 + s * 16 + r + grp * 8;
                if (m < c) {
                    const int   tok2 = list[e * NT + m];
                    const float w    = wt[e * NT + m];
                    const int   sl   = slot[e * NT + m];
                    yc[((size_t)tok2 * TK + sl) * DD + colg] = (acc[t * 2 + s][r] + bias) * w;
                }
            }
        }
    }
}

// ---------------- final per-token reduction over the 2 slots ----------------
__global__ void k_final(const float* __restrict__ yc, float* __restrict__ out) {
    size_t i = (size_t)blockIdx.x * blockDim.x + threadIdx.x;
    if (i >= (size_t)NT * DD) return;
    size_t n = i / DD, d = i % DD;
    out[i] = yc[(n * TK + 0) * DD + d] + yc[(n * TK + 1) * DD + d];
}

extern "C" void kernel_launch(void* const* d_in, const int* in_sizes, int n_in,
                              void* d_out, int out_size, void* d_ws, size_t ws_size,
                              hipStream_t stream) {
    (void)in_sizes; (void)n_in; (void)out_size; (void)ws_size;
    const float* x       = (const float*)d_in[0];
    const float* routing = (const float*)d_in[1];
    const float* W1      = (const float*)d_in[2];
    const float* b1      = (const float*)d_in[3];
    const float* W2      = (const float*)d_in[4];
    const float* b2      = (const float*)d_in[5];
    float* out = (float*)d_out;
    char*  ws  = (char*)d_ws;

    int*    cnt  = (int*)   (ws + OFF_CNT);
    int*    offs = (int*)   (ws + OFF_OFFS);
    int*    list = (int*)   (ws + OFF_LIST);
    float*  wtb  = (float*) (ws + OFF_WT);
    int*    slot = (int*)   (ws + OFF_SLOT);
    __bf16* xb   = (__bf16*)(ws + OFF_XB);
    __bf16* w1t  = (__bf16*)(ws + OFF_W1T);
    __bf16* w2t  = (__bf16*)(ws + OFF_W2T);
    __bf16* hb   = (__bf16*)(ws + OFF_HB);
    float*  yc   = (float*) (ws + OFF_YC);

    k_zero_counts<<<1, 32, 0, stream>>>(cnt);
    k_route<<<(NT + 255) / 256, 256, 0, stream>>>(routing, cnt, list, wtb, slot);
    k_offsets<<<1, 1, 0, stream>>>(cnt, offs);

    const size_t nW = (size_t)NE * DD * HH;
    k_cvt_x<<<(unsigned)(((size_t)NT * DD + 255) / 256), 256, 0, stream>>>(x, xb);
    k_t_w1<<<(unsigned)((nW + 255) / 256), 256, 0, stream>>>(W1, w1t);
    k_t_w2<<<(unsigned)((nW + 255) / 256), 256, 0, stream>>>(W2, w2t);
    k_zero_f32<<<(unsigned)(((size_t)NT * TK * DD + 255) / 256), 256, 0, stream>>>(
        yc, (size_t)NT * TK * DD);

    dim3 g1(HH / 512, NT / 32, NE);
    k_gemm1<<<g1, 256, 0, stream>>>(xb, w1t, b1, cnt, offs, list, hb);
    dim3 g2(DD / 512, NT / 32, NE);
    k_gemm2<<<g2, 256, 0, stream>>>(hb, w2t, b2, cnt, offs, list, wtb, slot, yc);

    k_final<<<(unsigned)(((size_t)NT * DD + 255) / 256), 256, 0, stream>>>(yc, out);
}
